// sigformer_5231270167237
// MI455X (gfx1250) — matrix operand order repaired and checked by reading, not compiled, not run on hardware
//
#include <hip/hip_runtime.h>

typedef __attribute__((ext_vector_type(16))) __bf16 v16bf;
typedef __attribute__((ext_vector_type(8)))  float  v8f;

// ---------------------------------------------------------------- utilities
__device__ __forceinline__ unsigned short f2bf(float f) {
    unsigned u = __float_as_uint(f);
    unsigned r = u + 0x7FFFu + ((u >> 16) & 1u);   // round-to-nearest-even
    return (unsigned short)(r >> 16);
}
__device__ __forceinline__ unsigned pack2(float lo, float hi) {
    return (unsigned)f2bf(lo) | ((unsigned)f2bf(hi) << 16);
}

union Frag { v16bf v; unsigned u[8]; };

// ---------------------------------------------------------------- kernel 1
// conv1d(SAME,k=3) + path assembly: path[b][t][0..2]=conv, [3]=x, [4]=t/99
__global__ void build_path(const float* __restrict__ x,
                           const float* __restrict__ cw,
                           const float* __restrict__ cb,
                           float* __restrict__ path) {
    int b = blockIdx.x, t = threadIdx.x;
    if (t < 100) {
        float xm = (t > 0)  ? x[b * 100 + t - 1] : 0.f;
        float x0 = x[b * 100 + t];
        float xp = (t < 99) ? x[b * 100 + t + 1] : 0.f;
        float* p = &path[(size_t)(b * 100 + t) * 5];
        #pragma unroll
        for (int o = 0; o < 3; ++o)
            p[o] = cw[o * 3] * xm + cw[o * 3 + 1] * x0 + cw[o * 3 + 2] * xp + cb[o];
        p[3] = x0;
        p[4] = (float)t * (1.f / 99.f);
    }
}

// ---------------------------------------------------------------- kernel 2
// depth-5 signature prefixes via incremental Chen composition.
__global__ __launch_bounds__(256) void sig_kernel(const float* __restrict__ path,
                                                  float* __restrict__ sig) {
    __shared__ float s1[5], s2[25], s3[125], s4[625], s5[3125];
    __shared__ float d1s[5], d2s[25], d3s[125], d4s[625];
    int b = blockIdx.x, tid = threadIdx.x;

    for (int e = tid; e < 3905; e += 256) {
        if (e < 5) s1[e] = 0.f; else if (e < 30) s2[e-5] = 0.f;
        else if (e < 155) s3[e-30] = 0.f; else if (e < 780) s4[e-155] = 0.f;
        else s5[e-780] = 0.f;
    }
    __syncthreads();

    for (int step = 0; step < 99; ++step) {
        if (tid < 5)
            d1s[tid] = path[(size_t)(b*100 + step + 1)*5 + tid]
                     - path[(size_t)(b*100 + step)*5 + tid];
        __syncthreads();
        float dl[5];
        #pragma unroll
        for (int c = 0; c < 5; ++c) dl[c] = d1s[c];
        for (int e = tid; e < 25; e += 256) d2s[e] = dl[e/5]*dl[e%5]*0.5f;
        for (int e = tid; e < 125; e += 256)
            d3s[e] = dl[e/25]*dl[(e/5)%5]*dl[e%5]*(1.f/6.f);
        for (int e = tid; e < 625; e += 256)
            d4s[e] = dl[e/125]*dl[(e/25)%5]*dl[(e/5)%5]*dl[e%5]*(1.f/24.f);
        __syncthreads();

        float nv[16];
        int cnt = 0;
        for (int f = tid; f < 3905; f += 256, ++cnt) {
            float v;
            if (f < 5) {
                v = s1[f] + dl[f];
            } else if (f < 30) { int g = f - 5;
                v = s2[g] + s1[g/5]*dl[g%5] + d2s[g];
            } else if (f < 155) { int g = f - 30;
                v = s3[g] + s2[g/5]*dl[g%5] + s1[g/25]*d2s[g%25] + d3s[g];
            } else if (f < 780) { int g = f - 155;
                v = s4[g] + s3[g/5]*dl[g%5] + s2[g/25]*d2s[g%25]
                  + s1[g/125]*d3s[g%125] + d4s[g];
            } else { int g = f - 780;
                v = s5[g] + s4[g/5]*dl[g%5] + s3[g/25]*d2s[g%25]
                  + s2[g/125]*d3s[g%125] + s1[g/625]*d4s[g%625]
                  + d4s[g/5]*dl[g%5]*0.2f;
            }
            nv[cnt] = v;
        }
        __syncthreads();

        bool emit = (step >= 8) && (((step - 8) % 10) == 0);
        float* out = emit ? &sig[((size_t)b*10 + (step - 8)/10) * 3905] : nullptr;
        cnt = 0;
        for (int f = tid; f < 3905; f += 256, ++cnt) {
            float v = nv[cnt];
            if (f < 5) s1[f] = v; else if (f < 30) s2[f-5] = v;
            else if (f < 155) s3[f-30] = v; else if (f < 780) s4[f-155] = v;
            else s5[f-780] = v;
            if (emit) out[f] = v;
        }
        __syncthreads();
    }
}

// ---------------------------------------------------------------- kernel 3
// C[M,N] = A[M,K] @ B[K,N] + bias, fp32 in/out, bf16 WMMA compute.
// Block tile 128x128, BK=32; 8 waves in 2x4 grid, each 64x32 (4x2 WMMA tiles).
// Double-buffered LDS: global loads for tile kt+1 issued before the WMMAs of
// tile kt; branch-free staging on interior tiles (guards hoisted).
#define BM 128
#define BN 128
#define BK 32
#define LDT 36   // LDS row stride (bf16 elems), keeps 4B alignment for pair reads

__global__ __launch_bounds__(256) void gemm_bf16(
        const float* __restrict__ A, int lda, int aoff,
        const float* __restrict__ B, int ldb,
        const float* __restrict__ bias,
        float* __restrict__ C, int ldc, int coff,
        int M, int N, int K) {
    __shared__ unsigned short As[2][BM * LDT];   // As[buf][m][k]
    __shared__ unsigned short Bs[2][BN * LDT];   // Bs[buf][n][k] (transposed)
    int tid  = threadIdx.x;
    int lane = tid & 31;
    int w    = tid >> 5;
    int wm   = w & 1;        // wave row (0..1)  -> 64 rows
    int wn   = w >> 1;       // wave col (0..3)  -> 32 cols
    int lh   = lane >> 4;    // lane half
    int ln   = lane & 15;
    int bm0  = blockIdx.x * BM;
    int bn0  = blockIdx.y * BN;
    bool fullM = (bm0 + BM <= M);
    bool fullN = (bn0 + BN <= N);

    // staging maps (pair of consecutive k per thread -> one b32 LDS store)
    // A: pair p = tid + 256*i (i<8): r = (tid>>4)+16i, k2 = (tid&15)*2
    // B: pair p = tid + 256*i (i<8): kp = (tid>>7)+2i, n  = tid&127
    int a_r0 = tid >> 4, a_k2 = (tid & 15) * 2;
    int b_k0 = tid >> 7, b_n  = tid & 127;

    float aLo[8], aHi[8], bLo[8], bHi[8];

    auto loadA = [&](int kt) {
        int k0 = kt * BK;
        if (fullM && (k0 + BK <= K)) {
            const float* base = A + (size_t)bm0 * lda + aoff + k0 + a_k2;
            #pragma unroll
            for (int i = 0; i < 8; ++i) {
                const float* p = base + (size_t)(a_r0 + 16 * i) * lda;
                aLo[i] = p[0]; aHi[i] = p[1];
            }
        } else {
            #pragma unroll
            for (int i = 0; i < 8; ++i) {
                int gr = bm0 + a_r0 + 16 * i;
                int gk = k0 + a_k2;
                const float* p = A + (size_t)gr * lda + aoff + gk;
                aLo[i] = (gr < M && gk     < K) ? p[0] : 0.f;
                aHi[i] = (gr < M && gk + 1 < K) ? p[1] : 0.f;
            }
        }
    };
    auto loadB = [&](int kt) {
        int k0 = kt * BK;
        if (fullN && (k0 + BK <= K)) {
            const float* base = B + (size_t)k0 * ldb + bn0 + b_n;
            #pragma unroll
            for (int i = 0; i < 8; ++i) {
                const float* p = base + (size_t)(2 * (b_k0 + 2 * i)) * ldb;
                bLo[i] = p[0]; bHi[i] = p[ldb];
            }
        } else {
            #pragma unroll
            for (int i = 0; i < 8; ++i) {
                int gk = k0 + 2 * (b_k0 + 2 * i);
                int gn = bn0 + b_n;
                const float* p = B + (size_t)gk * ldb + gn;
                bLo[i] = (gn < N && gk     < K) ? p[0]   : 0.f;
                bHi[i] = (gn < N && gk + 1 < K) ? p[ldb] : 0.f;
            }
        }
    };
    auto store = [&](int buf) {
        #pragma unroll
        for (int i = 0; i < 8; ++i)
            *(unsigned*)&As[buf][(a_r0 + 16 * i) * LDT + a_k2] = pack2(aLo[i], aHi[i]);
        #pragma unroll
        for (int i = 0; i < 8; ++i)
            *(unsigned*)&Bs[buf][b_n * LDT + 2 * (b_k0 + 2 * i)] = pack2(bLo[i], bHi[i]);
    };

    v8f zero = {0.f,0.f,0.f,0.f,0.f,0.f,0.f,0.f};
    v8f acc[4][2];
    #pragma unroll
    for (int i = 0; i < 4; ++i)
        #pragma unroll
        for (int j = 0; j < 2; ++j) acc[i][j] = zero;

    int nk = (K + BK - 1) / BK;

    loadA(0); loadB(0);
    store(0);
    __syncthreads();

    for (int kt = 0; kt < nk; ++kt) {
        int cur = kt & 1;
        if (kt + 1 < nk) { loadA(kt + 1); loadB(kt + 1); }  // overlap with WMMA

        // B fragments: lanes 0-15 -> K=0..15, lanes 16-31 -> K=16..31
        Frag bfr[2];
        #pragma unroll
        for (int j = 0; j < 2; ++j) {
            int n = wn * 32 + j * 16 + ln;
            const unsigned* p = (const unsigned*)&Bs[cur][n * LDT + lh * 16];
            #pragma unroll
            for (int pp = 0; pp < 8; ++pp) bfr[j].u[pp] = p[pp];
        }
        #pragma unroll
        for (int i = 0; i < 4; ++i) {
            // A fragment: pairs {0,2,4,...,14} (+16 for upper lane half)
            int m = wm * 64 + i * 16 + ln;
            const unsigned short* row = &As[cur][m * LDT];
            int kb = lh * 16;
            Frag afr;
            #pragma unroll
            for (int pp = 0; pp < 8; ++pp) {
                int kk = kb + 2 * pp;
                afr.u[pp] = *(const unsigned*)&row[kk];
            }
            #pragma unroll
            for (int j = 0; j < 2; ++j) {
                acc[i][j] = __builtin_amdgcn_wmma_f32_16x16x32_bf16(
                    false, afr.v, false, bfr[j].v, (short)0, acc[i][j],
                    false, false);
            }
        }

        if (kt + 1 < nk) store(cur ^ 1);
        __syncthreads();
    }

    // C/D layout: VGPR r, lanes 0-15: (M=r, N=lane); lanes 16-31: (M=8+r)
    #pragma unroll
    for (int i = 0; i < 4; ++i)
        #pragma unroll
        for (int j = 0; j < 2; ++j) {
            int col  = bn0 + wn * 32 + j * 16 + ln;
            int rowb = bm0 + wm * 64 + i * 16 + lh * 8;
            if (col < N) {
                float bv = bias ? bias[col] : 0.f;
                #pragma unroll
                for (int r = 0; r < 8; ++r) {
                    int row = rowb + r;
                    if (row < M)
                        C[(size_t)row * ldc + coff + col] = acc[i][j][r] + bv;
                }
            }
        }
}

// ---------------------------------------------------------------- kernel 4
// per-(batch, head) attention over seq=10
__global__ __launch_bounds__(256) void attn_kernel(const float* __restrict__ Q,
                                                   const float* __restrict__ Kb,
                                                   const float* __restrict__ V,
                                                   float* __restrict__ att) {
    __shared__ float S[10][10];
    const int dims[5] = {5, 25, 125, 625, 3125};
    const int offs[5] = {0, 5, 30, 155, 780};
    int b = blockIdx.x, h = blockIdx.y, tid = threadIdx.x;
    int d = dims[h], off = offs[h];

    if (tid < 100) {
        int q = tid / 10, kk = tid % 10;
        const float* qp = &Q[((size_t)b*10 + q) * 3905 + off];
        const float* kp = &Kb[((size_t)b*10 + kk) * 3905 + off];
        float acc = 0.f;
        for (int j = 0; j < d; ++j) acc += qp[j] * kp[j];
        S[q][kk] = acc * rsqrtf((float)d);
    }
    __syncthreads();
    if (tid < 10) {
        float mx = -1e30f;
        #pragma unroll
        for (int k = 0; k < 10; ++k) mx = fmaxf(mx, S[tid][k]);
        float e[10], sum = 0.f;
        #pragma unroll
        for (int k = 0; k < 10; ++k) { e[k] = __expf(S[tid][k] - mx); sum += e[k]; }
        float inv = 1.f / sum;
        #pragma unroll
        for (int k = 0; k < 10; ++k) S[tid][k] = e[k] * inv;
    }
    __syncthreads();
    for (int j = tid; j < d; j += 256) {
        float vv[10];
        #pragma unroll
        for (int k = 0; k < 10; ++k)
            vv[k] = V[((size_t)b*10 + k) * 3905 + off + j];
        #pragma unroll
        for (int q = 0; q < 10; ++q) {
            float acc = 0.f;
            #pragma unroll
            for (int k = 0; k < 10; ++k) acc += S[q][k] * vv[k];
            att[((size_t)b*10 + q) * 3905 + off + j] = acc;
        }
    }
}

// ---------------------------------------------------------------- kernel 5
// 39050 -> 32 -> 32 -> 32 -> 32 -> 32 -> 1, relu/sigmoid. One block per row.
__global__ __launch_bounds__(256) void mlp_kernel(const float* __restrict__ H,
        const float* __restrict__ W0, const float* __restrict__ B0,
        const float* __restrict__ W1, const float* __restrict__ B1,
        const float* __restrict__ W2, const float* __restrict__ B2,
        const float* __restrict__ W3, const float* __restrict__ B3,
        const float* __restrict__ W4, const float* __restrict__ B4,
        const float* __restrict__ W5, const float* __restrict__ B5,
        float* __restrict__ out) {
    __shared__ float part[8][32];
    __shared__ float ha[32], hb[32];
    int b = blockIdx.x, tid = threadIdx.x;
    int n = tid & 31, c = tid >> 5;
    const float* hrow = &H[(size_t)b * 39050];
    float acc = 0.f;
    for (int k = c; k < 39050; k += 8) acc += hrow[k] * W0[(size_t)k * 32 + n];
    part[c][n] = acc;
    __syncthreads();
    if (tid < 32) {
        float s = B0[tid];
        #pragma unroll
        for (int cc = 0; cc < 8; ++cc) s += part[cc][tid];
        ha[tid] = fmaxf(s, 0.f);
    }
    __syncthreads();
    const float* Ws[4]  = {W1, W2, W3, W4};
    const float* Bs2[4] = {B1, B2, B3, B4};
    for (int ly = 0; ly < 4; ++ly) {
        if (tid < 32) {
            float s = Bs2[ly][tid];
            #pragma unroll
            for (int j = 0; j < 32; ++j) s += ha[j] * Ws[ly][j * 32 + tid];
            hb[tid] = fmaxf(s, 0.f);
        }
        __syncthreads();
        if (tid < 32) ha[tid] = hb[tid];
        __syncthreads();
    }
    if (tid == 0) {
        float s = B5[0];
        #pragma unroll
        for (int j = 0; j < 32; ++j) s += ha[j] * W5[j];
        out[b] = 1.f / (1.f + __expf(-s));
    }
}

// ---------------------------------------------------------------- launcher
extern "C" void kernel_launch(void* const* d_in, const int* in_sizes, int n_in,
                              void* d_out, int out_size, void* d_ws, size_t ws_size,
                              hipStream_t stream) {
    (void)in_sizes; (void)n_in; (void)out_size; (void)ws_size;
    const float* x      = (const float*)d_in[0];
    const float* conv_w = (const float*)d_in[1];
    const float* conv_b = (const float*)d_in[2];
    const float* q_w[5]; const float* q_b[5];
    const float* k_w[5]; const float* k_b[5];
    const float* v_w[5]; const float* v_b[5];
    for (int h = 0; h < 5; ++h) {
        q_w[h] = (const float*)d_in[3 + h];   q_b[h] = (const float*)d_in[8 + h];
        k_w[h] = (const float*)d_in[13 + h];  k_b[h] = (const float*)d_in[18 + h];
        v_w[h] = (const float*)d_in[23 + h];  v_b[h] = (const float*)d_in[28 + h];
    }
    const float* o_w = (const float*)d_in[33];
    const float* o_b = (const float*)d_in[34];
    const float* dw[6]; const float* db[6];
    for (int i = 0; i < 6; ++i) {
        dw[i] = (const float*)d_in[35 + i];
        db[i] = (const float*)d_in[41 + i];
    }

    // workspace layout (floats): path | sig | Q | K | V
    // (sig reused as att, Q reused as the post-projection hidden state)
    const size_t ROWS = 1280, SD = 3905;
    float* ws   = (float*)d_ws;
    float* path = ws;                       // 128*100*5 = 64000
    float* sig  = path + 64000;             // ROWS*SD
    float* Qb   = sig + ROWS * SD;
    float* Kbuf = Qb  + ROWS * SD;
    float* Vb   = Kbuf + ROWS * SD;
    float* att  = sig;                      // reuse: sig dead after QKV GEMMs
    float* hbuf = Qb;                       // reuse: Q dead after attention

    build_path<<<dim3(128), dim3(128), 0, stream>>>(x, conv_w, conv_b, path);
    sig_kernel<<<dim3(128), dim3(256), 0, stream>>>(path, sig);

    const int dims[5] = {5, 25, 125, 625, 3125};
    const int offs[5] = {0, 5, 30, 155, 780};
    for (int h = 0; h < 5; ++h) {
        int d = dims[h], off = offs[h];
        dim3 g((unsigned)((ROWS + BM - 1) / BM), (unsigned)((d + BN - 1) / BN));
        gemm_bf16<<<g, 256, 0, stream>>>(sig, 3905, off, q_w[h], d, q_b[h],
                                         Qb, 3905, off, (int)ROWS, d, d);
        gemm_bf16<<<g, 256, 0, stream>>>(sig, 3905, off, k_w[h], d, k_b[h],
                                         Kbuf, 3905, off, (int)ROWS, d, d);
        gemm_bf16<<<g, 256, 0, stream>>>(sig, 3905, off, v_w[h], d, v_b[h],
                                         Vb, 3905, off, (int)ROWS, d, d);
    }

    attn_kernel<<<dim3(128, 5), dim3(256), 0, stream>>>(Qb, Kbuf, Vb, att);

    {
        dim3 g((unsigned)((ROWS + BM - 1) / BM), (unsigned)((3905 + BN - 1) / BN));
        gemm_bf16<<<g, 256, 0, stream>>>(att, 3905, 0, o_w, 3905, o_b,
                                         hbuf, 3905, 0, (int)ROWS, 3905, 3905);
    }

    mlp_kernel<<<dim3(128), dim3(256), 0, stream>>>(hbuf,
        dw[0], db[0], dw[1], db[1], dw[2], db[2],
        dw[3], db[3], dw[4], db[4], dw[5], db[5],
        (float*)d_out);
}
